// MultiHeadAttention_68736656605864
// MI455X (gfx1250) — matrix operand order repaired and checked by reading, not compiled, not run on hardware
//
#include <hip/hip_runtime.h>

// ---------------------------------------------------------------------------
// T5-style MHA with relative position bias, MI455X (gfx1250, wave32, WMMA).
// B=4, T=2048, D=768, H=12, DK=64.
// Flash-attention core: 8 waves/WG, K/V chunks staged in LDS via the Tensor
// Data Mover (double-buffered), f16 WMMA everywhere, fp32 accumulate.
// ---------------------------------------------------------------------------

namespace {
constexpr int kB   = 4;
constexpr int kT   = 2048;
constexpr int kD   = 768;
constexpr int kH   = 12;
constexpr int kDK  = 64;
constexpr int kLUTW = 2 * kT - 1;   // rel in [-2047, 2047]
constexpr int kWaves = 8;           // waves per attention workgroup
constexpr int kKStride = 72;        // 64 halves + TDM pad (8 halves)
constexpr int kVStride = 40;        // 32 halves + TDM pad (8 halves)
}

typedef __attribute__((ext_vector_type(16))) _Float16     v16h;
typedef __attribute__((ext_vector_type(8)))  _Float16     v8h;
typedef __attribute__((ext_vector_type(8)))  float        v8f;
typedef __attribute__((ext_vector_type(4)))  float        v4f;
typedef __attribute__((ext_vector_type(4)))  unsigned int v4u;
typedef __attribute__((ext_vector_type(8)))  int          v8i;
typedef __attribute__((ext_vector_type(4)))  int          v4i;

__device__ __forceinline__ v8f wmma_f16(v16h a, v16h b, v8f c) {
  // D = A(16x32 f16) * B(32x16 f16) + C(16x16 f32)
  return __builtin_amdgcn_wmma_f32_16x16x32_f16(
      /*neg_a=*/false, a, /*neg_b=*/false, b,
      /*c_mod=*/(short)0, c, /*reuse_a=*/false, /*reuse_b=*/false);
}

// Low 32 bits of a flat LDS address are the LDS byte offset (aperture scheme).
__device__ __forceinline__ unsigned lds_off(const void* p) {
  return (unsigned)(uintptr_t)p;
}

// ---------------------------------------------------------------------------
// Tensor Data Mover: 2D tile (tile_d0 x tile_d1 halves, row stride stride0
// elements) from global to LDS, with LDS padding (bank de-conflict).
//   pad_interval code: threshold = 2^(code+1) DWORDs ; pad_amount code: code+1 DWORDs
// ---------------------------------------------------------------------------
__device__ __forceinline__ void tdm_load_2d(unsigned lds_addr, const void* gptr,
                                            unsigned tile_d0, unsigned tile_d1,
                                            unsigned long long stride0_elems,
                                            unsigned pad_interval,
                                            unsigned pad_amount) {
  const unsigned long long ga = (unsigned long long)(uintptr_t)gptr;
  const unsigned td0 = 0x40000000u;           // huge tensor dims: no OOB clipping
  const unsigned td1 = 0x40000000u;

  v4u g0;
  g0[0] = 1u;                                  // count=1, user-mode descriptor
  g0[1] = lds_addr;                            // D#.lds_addr
  g0[2] = (unsigned)ga;                        // global_addr[31:0]
  g0[3] = (unsigned)((ga >> 32) & 0x01ffffffu) | (2u << 30);  // ga[56:32] | type=2

  v8i g1;
  g1[0] = (int)((1u << 16)                     // data_size = 2 bytes
              | (1u << 20)                     // pad_enable
              | (pad_interval << 22)
              | (pad_amount << 25));
  g1[1] = (int)((td0 & 0xffffu) << 16);                        // tensor_dim0[15:0]
  g1[2] = (int)((td0 >> 16) | ((td1 & 0xffffu) << 16));        // td0[31:16] | td1[15:0]
  g1[3] = (int)((td1 >> 16) | ((tile_d0 & 0xffffu) << 16));    // td1[31:16] | tile_dim0
  g1[4] = (int)(tile_d1 & 0xffffu);                            // tile_dim1 (tile_dim2=0)
  g1[5] = (int)(unsigned)(stride0_elems & 0xffffffffu);        // dim0_stride[31:0]
  g1[6] = (int)(unsigned)((stride0_elems >> 32) & 0xffffu);    // dim0_stride[47:32]
  g1[7] = 0;

  v4i z4 = {0, 0, 0, 0};
#if defined(__clang_major__) && (__clang_major__ >= 23)
  v8i z8 = {0, 0, 0, 0, 0, 0, 0, 0};
  __builtin_amdgcn_tensor_load_to_lds(g0, g1, z4, z4, z8, 0);
#else
  __builtin_amdgcn_tensor_load_to_lds(g0, g1, z4, z4, 0);
#endif
}

// ---------------------------------------------------------------------------
// Kernel 0: bias LUT  lut[h][rel + (T-1)] = rel_table[bucket(rel)][h]
// ---------------------------------------------------------------------------
__global__ void k_bias_lut(const float* __restrict__ rel_table,
                           float* __restrict__ lut) {
  int idx = blockIdx.x * blockDim.x + threadIdx.x;
  if (idx >= kH * kLUTW) return;
  int h   = idx / kLUTW;
  int r   = idx - h * kLUTW;
  int rel = r - (kT - 1);                 // rel = k_pos - q_pos
  int n   = -rel;                         // reference: n = -rel_pos
  int ret = (n < 0) ? 16 : 0;             // NUM_BUCKETS/2
  int an  = (n < 0) ? -n : n;
  int bkt;
  if (an < 8) {
    bkt = an;
  } else {
    float v = __logf((float)an / 8.0f + 1e-6f) / __logf(128.0f / 8.0f) * 8.0f;
    int vi  = 8 + (int)v;                 // trunc toward zero, matches astype(i32)
    bkt = vi < 15 ? vi : 15;
  }
  lut[idx] = rel_table[(ret + bkt) * kH + h];
}

// ---------------------------------------------------------------------------
// Kernel 1: projection GEMM  C[m,n] = sum_k X[m,k] * W[n,k] + bias[n]
//   transposed==0 : out[((b*H+h)*T + t)*DK + dv]  (Q, K layout)
//   transposed==1 : out[((b*H+h)*DK + dv)*T + t]  (V^T layout)
// ---------------------------------------------------------------------------
__global__ void k_proj(const float* __restrict__ X, const float* __restrict__ W,
                       const float* __restrict__ bias, _Float16* __restrict__ out,
                       int transposed) {
  const int lane = threadIdx.x & 31;
  const int l16  = lane & 15;
  const int hiw  = lane >> 4;
  const int mt   = blockIdx.x;
  const int nt   = blockIdx.y;
  const int mrow = mt * 16 + l16;
  const int ncol = nt * 16 + l16;

  const float* arow = X + (size_t)mrow * kD;
  const float* brow = W + (size_t)ncol * kD;

  v8f c = {0.f, 0.f, 0.f, 0.f, 0.f, 0.f, 0.f, 0.f};

  for (int kb = 0; kb < kD; kb += 32) {
    if (kb + 32 < kD) {
      __builtin_prefetch(arow + kb + 32, 0, 0);
      __builtin_prefetch(brow + kb + 32, 0, 0);
    }
    v16h a, b;
    {
      const float* p = arow + kb + hiw * 16;
      v4f x0 = *(const v4f*)(p);
      v4f x1 = *(const v4f*)(p + 4);
      v4f x2 = *(const v4f*)(p + 8);
      v4f x3 = *(const v4f*)(p + 12);
#pragma unroll
      for (int j = 0; j < 4; ++j) {
        a[j]      = (_Float16)x0[j];
        a[4 + j]  = (_Float16)x1[j];
        a[8 + j]  = (_Float16)x2[j];
        a[12 + j] = (_Float16)x3[j];
      }
    }
    {
      const float* p = brow + kb + hiw * 16;
      v4f y0 = *(const v4f*)(p);
      v4f y1 = *(const v4f*)(p + 4);
      v4f y2 = *(const v4f*)(p + 8);
      v4f y3 = *(const v4f*)(p + 12);
#pragma unroll
      for (int j = 0; j < 4; ++j) {
        b[j]      = (_Float16)y0[j];
        b[4 + j]  = (_Float16)y1[j];
        b[8 + j]  = (_Float16)y2[j];
        b[12 + j] = (_Float16)y3[j];
      }
    }
    c = wmma_f16(a, b, c);
  }

  const float bn = bias[ncol];
  const int hh = ncol >> 6;
  const int dv = ncol & 63;
#pragma unroll
  for (int i = 0; i < 8; ++i) {
    int m    = mt * 16 + i + 8 * hiw;
    int bidx = m >> 11;
    int t    = m & (kT - 1);
    float val = c[i] + bn;
    size_t off;
    if (transposed) off = (((size_t)bidx * kH + hh) * kDK + dv) * kT + t;
    else            off = (((size_t)bidx * kH + hh) * kT + t) * kDK + dv;
    out[off] = (_Float16)val;
  }
}

// ---------------------------------------------------------------------------
// Kernel 2: flash attention. 8 waves/WG; each wave owns a 16-query tile.
// K/V 32-key chunks staged in LDS by the TDM (double buffered).
//   Q,K: [B,H,T,64] f16.  VT: [B,H,64,T] f16.  Xo: [B,T,H,64] f16.
// ---------------------------------------------------------------------------
__global__ __launch_bounds__(kWaves * 32)
void k_attn(const _Float16* __restrict__ Q,
            const _Float16* __restrict__ K,
            const _Float16* __restrict__ VT,
            const int* __restrict__ mask,
            const float* __restrict__ lut,
            _Float16* __restrict__ Xo) {
  const int tid  = threadIdx.x;
  const int lane = tid & 31;
  const int wid  = tid >> 5;
  const int l16  = lane & 15;
  const int hiw  = lane >> 4;
  const int qt = blockIdx.x * kWaves + wid;   // 16-query tile index, 0..127
  const int h  = blockIdx.y;
  const int b  = blockIdx.z;

  const size_t bh = (size_t)b * kH + h;
  const _Float16* Qb = Q  + bh * kT * kDK;
  const _Float16* Kb = K  + bh * kT * kDK;
  const _Float16* Vb = VT + bh * kDK * kT;
  const float* lh = lut + (size_t)h * kLUTW + (kT - 1);
  const int* mbase = mask + (size_t)b * kT * kT;

  // kbuf: 32 keys x 64 dk, TDM-padded to 72-half rows (conflict-free b128)
  // vbuf: 64 dv x 32 keys, TDM-padded to 40-half rows (conflict-free b128)
  __shared__ __align__(16) _Float16 kbuf[2][32 * kKStride + 8];
  __shared__ __align__(16) _Float16 vbuf[2][64 * kVStride + 8];
  __shared__ __align__(16) _Float16 pbuf[kWaves][16][40];

  // ---- TDM prologue: stage chunk 0 ----------------------------------------
  if (wid == 0) {
    // K chunk: 32 contiguous rows of 64 halves == one 2048-half line.
    tdm_load_2d(lds_off(&kbuf[0][0]), Kb, /*tile_d0=*/2048, /*tile_d1=*/1,
                /*stride0=*/2048, /*pad_interval=*/4, /*pad_amount=*/3);
    // V^T chunk: 64 rows of 32 halves, row stride T.
    tdm_load_2d(lds_off(&vbuf[0][0]), Vb, /*tile_d0=*/32, /*tile_d1=*/64,
                /*stride0=*/(unsigned long long)kT, /*pad_interval=*/3,
                /*pad_amount=*/3);
    __builtin_amdgcn_s_wait_tensorcnt(0);
  }
  __syncthreads();

  // ---- Q A-fragments, held in registers all loop long ---------------------
  v16h qa0, qa1;
  {
    const _Float16* p0 = Qb + (size_t)(qt * 16 + l16) * kDK + hiw * 16;
    v8h t0 = *(const v8h*)(p0);
    v8h t1 = *(const v8h*)(p0 + 8);
    v8h t2 = *(const v8h*)(p0 + 32);
    v8h t3 = *(const v8h*)(p0 + 40);
#pragma unroll
    for (int j = 0; j < 8; ++j) {
      qa0[j] = t0[j]; qa0[8 + j] = t1[j];
      qa1[j] = t2[j]; qa1[8 + j] = t3[j];
    }
  }

  v8f o0 = {0.f,0.f,0.f,0.f,0.f,0.f,0.f,0.f};
  v8f o1 = o0, o2 = o0, o3 = o0;
  float mrow[8], lrow[8];
#pragma unroll
  for (int i = 0; i < 8; ++i) { mrow[i] = -1e30f; lrow[i] = 0.f; }

  for (int j0 = 0; j0 < kT; j0 += 32) {
    const int cur = (j0 >> 5) & 1;

    // ---- issue TDM for next chunk into the other buffer -------------------
    if (wid == 0 && (j0 + 32) < kT) {
      tdm_load_2d(lds_off(&kbuf[cur ^ 1][0]), Kb + (size_t)(j0 + 32) * kDK,
                  2048, 1, 2048, 4, 3);
      tdm_load_2d(lds_off(&vbuf[cur ^ 1][0]), Vb + (j0 + 32),
                  32, 64, (unsigned long long)kT, 3, 3);
    }

    // ---- S = Q K^T for 32 keys: two 16x16 C tiles (B-frags from LDS) ------
    v8f s0 = {0.f,0.f,0.f,0.f,0.f,0.f,0.f,0.f};
    v8f s1 = s0;
    {
      const _Float16* kr = &kbuf[cur][(size_t)l16 * kKStride + hiw * 16];
      v8h t0 = *(const v8h*)(kr);
      v8h t1 = *(const v8h*)(kr + 8);
      v8h t2 = *(const v8h*)(kr + 32);
      v8h t3 = *(const v8h*)(kr + 40);
      v16h kb0, kb1;
#pragma unroll
      for (int j = 0; j < 8; ++j) {
        kb0[j] = t0[j]; kb0[8 + j] = t1[j];
        kb1[j] = t2[j]; kb1[8 + j] = t3[j];
      }
      s0 = wmma_f16(qa0, kb0, s0);
      s0 = wmma_f16(qa1, kb1, s0);
    }
    {
      const _Float16* kr = &kbuf[cur][(size_t)(16 + l16) * kKStride + hiw * 16];
      v8h t0 = *(const v8h*)(kr);
      v8h t1 = *(const v8h*)(kr + 8);
      v8h t2 = *(const v8h*)(kr + 32);
      v8h t3 = *(const v8h*)(kr + 40);
      v16h kb0, kb1;
#pragma unroll
      for (int j = 0; j < 8; ++j) {
        kb0[j] = t0[j]; kb0[8 + j] = t1[j];
        kb1[j] = t2[j]; kb1[8 + j] = t3[j];
      }
      s1 = wmma_f16(qa0, kb0, s1);
      s1 = wmma_f16(qa1, kb1, s1);
    }

    // ---- bias + mask + online softmax -------------------------------------
    float fac[8];
#pragma unroll
    for (int i = 0; i < 8; ++i) {
      const int qpos = qt * 16 + i + 8 * hiw;
      const int k0   = j0 + l16;
      const int k1   = k0 + 16;
      float sc0 = s0[i] * 0.125f + lh[k0 - qpos];
      float sc1 = s1[i] * 0.125f + lh[k1 - qpos];
      const int* mr = mbase + (size_t)qpos * kT;
      if (mr[k0] == 0) sc0 = -1e9f;
      if (mr[k1] == 0) sc1 = -1e9f;

      float v = fmaxf(sc0, sc1);
      v = fmaxf(v, __shfl_xor(v, 1, 32));
      v = fmaxf(v, __shfl_xor(v, 2, 32));
      v = fmaxf(v, __shfl_xor(v, 4, 32));
      v = fmaxf(v, __shfl_xor(v, 8, 32));
      const float mnew = fmaxf(mrow[i], v);
      const float f    = __expf(mrow[i] - mnew);
      const float p0   = __expf(sc0 - mnew);
      const float p1   = __expf(sc1 - mnew);
      float rs = p0 + p1;
      rs += __shfl_xor(rs, 1, 32);
      rs += __shfl_xor(rs, 2, 32);
      rs += __shfl_xor(rs, 4, 32);
      rs += __shfl_xor(rs, 8, 32);
      lrow[i] = lrow[i] * f + rs;
      mrow[i] = mnew;
      fac[i]  = f;
      pbuf[wid][i + 8 * hiw][l16]      = (_Float16)p0;
      pbuf[wid][i + 8 * hiw][16 + l16] = (_Float16)p1;
    }
#pragma unroll
    for (int i = 0; i < 8; ++i) {
      o0[i] *= fac[i]; o1[i] *= fac[i]; o2[i] *= fac[i]; o3[i] *= fac[i];
    }
    __syncthreads();

    // ---- P (16x32) as A-fragment via per-wave LDS swizzle -----------------
    v16h pa;
    {
      const _Float16* pr = &pbuf[wid][l16][hiw * 16];
      v8h t0 = *(const v8h*)(pr);
      v8h t1 = *(const v8h*)(pr + 8);
#pragma unroll
      for (int j = 0; j < 8; ++j) { pa[j] = t0[j]; pa[8 + j] = t1[j]; }
    }

    // ---- O += P * V : 4 dv tiles, B-frags from LDS ------------------------
    {
      const _Float16* vr = &vbuf[cur][(size_t)(0 * 16 + l16) * kVStride + hiw * 16];
      v8h t0 = *(const v8h*)(vr); v8h t1 = *(const v8h*)(vr + 8);
      v16h vb;
#pragma unroll
      for (int j = 0; j < 8; ++j) { vb[j] = t0[j]; vb[8 + j] = t1[j]; }
      o0 = wmma_f16(pa, vb, o0);
    }
    {
      const _Float16* vr = &vbuf[cur][(size_t)(1 * 16 + l16) * kVStride + hiw * 16];
      v8h t0 = *(const v8h*)(vr); v8h t1 = *(const v8h*)(vr + 8);
      v16h vb;
#pragma unroll
      for (int j = 0; j < 8; ++j) { vb[j] = t0[j]; vb[8 + j] = t1[j]; }
      o1 = wmma_f16(pa, vb, o1);
    }
    {
      const _Float16* vr = &vbuf[cur][(size_t)(2 * 16 + l16) * kVStride + hiw * 16];
      v8h t0 = *(const v8h*)(vr); v8h t1 = *(const v8h*)(vr + 8);
      v16h vb;
#pragma unroll
      for (int j = 0; j < 8; ++j) { vb[j] = t0[j]; vb[8 + j] = t1[j]; }
      o2 = wmma_f16(pa, vb, o2);
    }
    {
      const _Float16* vr = &vbuf[cur][(size_t)(3 * 16 + l16) * kVStride + hiw * 16];
      v8h t0 = *(const v8h*)(vr); v8h t1 = *(const v8h*)(vr + 8);
      v16h vb;
#pragma unroll
      for (int j = 0; j < 8; ++j) { vb[j] = t0[j]; vb[8 + j] = t1[j]; }
      o3 = wmma_f16(pa, vb, o3);
    }

    // ---- close the iteration: DMA for next chunk must have landed ---------
    if (wid == 0) __builtin_amdgcn_s_wait_tensorcnt(0);
    __syncthreads();
  }

  // ---- normalize and store Xo[b, t, h, dv] --------------------------------
#pragma unroll
  for (int i = 0; i < 8; ++i) {
    const int qpos = qt * 16 + i + 8 * hiw;
    const float inv = 1.0f / lrow[i];
    _Float16* dst = Xo + (((size_t)b * kT + qpos) * kH + h) * kDK;
    dst[0  + l16] = (_Float16)(o0[i] * inv);
    dst[16 + l16] = (_Float16)(o1[i] * inv);
    dst[32 + l16] = (_Float16)(o2[i] * inv);
    dst[48 + l16] = (_Float16)(o3[i] * inv);
  }
}

// ---------------------------------------------------------------------------
// Kernel 3: output projection  out[m,n] = sum_k Xa[m,k]*Wo[n,k] + bo[n]  (f32)
// ---------------------------------------------------------------------------
__global__ void k_outproj(const _Float16* __restrict__ Xa,
                          const float* __restrict__ Wo,
                          const float* __restrict__ bo,
                          float* __restrict__ out) {
  const int lane = threadIdx.x & 31;
  const int l16  = lane & 15;
  const int hiw  = lane >> 4;
  const int mt   = blockIdx.x;
  const int nt   = blockIdx.y;
  const int mrow = mt * 16 + l16;
  const int ncol = nt * 16 + l16;

  const _Float16* arow = Xa + (size_t)mrow * kD;
  const float*    brow = Wo + (size_t)ncol * kD;

  v8f c = {0.f, 0.f, 0.f, 0.f, 0.f, 0.f, 0.f, 0.f};
  for (int kb = 0; kb < kD; kb += 32) {
    if (kb + 32 < kD) {
      __builtin_prefetch(arow + kb + 32, 0, 0);
      __builtin_prefetch(brow + kb + 32, 0, 0);
    }
    v16h a, b;
    {
      const _Float16* p = arow + kb + hiw * 16;
      v8h t0 = *(const v8h*)(p);
      v8h t1 = *(const v8h*)(p + 8);
#pragma unroll
      for (int j = 0; j < 8; ++j) { a[j] = t0[j]; a[8 + j] = t1[j]; }
    }
    {
      const float* p = brow + kb + hiw * 16;
      v4f y0 = *(const v4f*)(p);
      v4f y1 = *(const v4f*)(p + 4);
      v4f y2 = *(const v4f*)(p + 8);
      v4f y3 = *(const v4f*)(p + 12);
#pragma unroll
      for (int j = 0; j < 4; ++j) {
        b[j]      = (_Float16)y0[j];
        b[4 + j]  = (_Float16)y1[j];
        b[8 + j]  = (_Float16)y2[j];
        b[12 + j] = (_Float16)y3[j];
      }
    }
    c = wmma_f16(a, b, c);
  }

  const float bn = bo[ncol];
#pragma unroll
  for (int i = 0; i < 8; ++i) {
    int m = mt * 16 + i + 8 * hiw;
    out[(size_t)m * kD + ncol] = c[i] + bn;
  }
}

// ---------------------------------------------------------------------------
// Launch
// ---------------------------------------------------------------------------
extern "C" void kernel_launch(void* const* d_in, const int* in_sizes, int n_in,
                              void* d_out, int out_size, void* d_ws, size_t ws_size,
                              hipStream_t stream) {
  const float* query = (const float*)d_in[0];
  const float* key   = (const float*)d_in[1];
  const float* value = (const float*)d_in[2];
  const int*   mask  = (const int*)  d_in[3];
  const float* Wq    = (const float*)d_in[4];
  const float* bq    = (const float*)d_in[5];
  const float* Wk    = (const float*)d_in[6];
  const float* bk    = (const float*)d_in[7];
  const float* Wv    = (const float*)d_in[8];
  const float* bv    = (const float*)d_in[9];
  const float* Wo    = (const float*)d_in[10];
  const float* bo    = (const float*)d_in[11];
  const float* rel_table = (const float*)d_in[12];

  const size_t qkvElems = (size_t)kB * kH * kT * kDK;   // 6,291,456 halves
  char* ws = (char*)d_ws;
  _Float16* qbuf = (_Float16*)ws;  ws += qkvElems * sizeof(_Float16);
  _Float16* kbuf = (_Float16*)ws;  ws += qkvElems * sizeof(_Float16);
  _Float16* vtbf = (_Float16*)ws;  ws += qkvElems * sizeof(_Float16);
  _Float16* xabf = (_Float16*)ws;  ws += qkvElems * sizeof(_Float16);
  float*    lut  = (float*)ws;

  {
    int total = kH * kLUTW;
    k_bias_lut<<<dim3((total + 255) / 256), dim3(256), 0, stream>>>(rel_table, lut);
  }
  {
    dim3 grid((kB * kT) / 16, kD / 16);
    k_proj<<<grid, dim3(32), 0, stream>>>(query, Wq, bq, qbuf, 0);
    k_proj<<<grid, dim3(32), 0, stream>>>(key,   Wk, bk, kbuf, 0);
    k_proj<<<grid, dim3(32), 0, stream>>>(value, Wv, bv, vtbf, 1);
  }
  k_attn<<<dim3(kT / (16 * kWaves), kH, kB), dim3(kWaves * 32), 0, stream>>>(
      qbuf, kbuf, vtbf, mask, lut, xabf);
  k_outproj<<<dim3((kB * kT) / 16, kD / 16), dim3(32), 0, stream>>>(
      xabf, Wo, bo, (float*)d_out);
}
